// AttnDecoderRNN_68332929679463
// MI455X (gfx1250) — matrix operand order repaired and checked by reading, not compiled, not run on hardware
//
#include <hip/hip_runtime.h>
#include <hip/hip_bf16.h>

#define HDIM 128
#define VOCAB 32000
#define BATCH 32
#define SLEN 256
#define TLEN 128

typedef __attribute__((ext_vector_type(16))) __bf16 bf16x16;
typedef __attribute__((ext_vector_type(8)))  float  floatx8;

__device__ __forceinline__ float sigmoidf_(float x) { return 1.0f / (1.0f + __expf(-x)); }

// ---------------------------------------------------------------------------
// One-time helpers
// ---------------------------------------------------------------------------
__global__ void k_f32_to_bf16(const float* __restrict__ src, __bf16* __restrict__ dst, int n) {
    int i = blockIdx.x * blockDim.x + threadIdx.x;
    if (i < n) dst[i] = (__bf16)src[i];
}

__global__ void k_zero_f32(float* __restrict__ p, int n) {
    int i = blockIdx.x * blockDim.x + threadIdx.x;
    if (i < n) p[i] = 0.0f;
}

__global__ void k_copy_f32(const float* __restrict__ src, float* __restrict__ dst, int n) {
    int i = blockIdx.x * blockDim.x + threadIdx.x;
    if (i < n) dst[i] = src[i];
}

// uk[b,s,:] = enc[b,s,:] @ Ua_w.T + Ua_b   (one block per (b,s) row, f32)
__global__ void k_uk(const float* __restrict__ enc, const float* __restrict__ Ua_w,
                     const float* __restrict__ Ua_b, float* __restrict__ uk) {
    __shared__ float row[HDIM];
    const int bs = blockIdx.x;       // b*SLEN + s
    const int i  = threadIdx.x;      // 0..127
    row[i] = enc[(long long)bs * HDIM + i];
    __syncthreads();
    float acc = Ua_b[i];
    const float* wr = Ua_w + i * HDIM;
#pragma unroll 8
    for (int k = 0; k < HDIM; ++k) acc = fmaf(wr[k], row[k], acc);
    uk[(long long)bs * HDIM + i] = acc;
}

// ---------------------------------------------------------------------------
// Per-step attention + GRU cell. One block per batch element, 128 threads.
// ---------------------------------------------------------------------------
__global__ void k_step(const float* __restrict__ enc, const float* __restrict__ uk,
                       const int*   __restrict__ tgt, const float* __restrict__ emb,
                       const float* __restrict__ Wa_w, const float* __restrict__ Wa_b,
                       const float* __restrict__ Va_w, const float* __restrict__ Va_b,
                       const float* __restrict__ W_ih, const float* __restrict__ b_ih,
                       const float* __restrict__ W_hh, const float* __restrict__ b_hh,
                       const float* __restrict__ h_prev, float* __restrict__ h_next,
                       __bf16* __restrict__ h_bf16,
                       float* __restrict__ attn_out,   // non-null only on last step
                       float* __restrict__ hfin_out,   // non-null only on last step
                       int t) {
    __shared__ float q[HDIM], wq[HDIM], xs[2 * HDIM], wsm[SLEN], red[HDIM];
    const int b = blockIdx.x;
    const int i = threadIdx.x;

    const float hi = h_prev[b * HDIM + i];
    q[i] = hi;
    __syncthreads();

    // wq = Wa_w @ q + Wa_b
    {
        float acc = Wa_b[i];
        const float* wr = Wa_w + i * HDIM;
#pragma unroll 8
        for (int k = 0; k < HDIM; ++k) acc = fmaf(wr[k], q[k], acc);
        wq[i] = acc;
    }
    __syncthreads();

    // scores[s] = Va . tanh(wq + uk[b,s,:]) + Va_b   (2 s-values per thread)
    const float vab = Va_b[0];
    float sc0, sc1;
    {
        const float* u0 = uk + ((long long)b * SLEN + i) * HDIM;
        const float* u1 = u0 + 128 * HDIM;
        float a0 = 0.0f, a1 = 0.0f;
#pragma unroll 4
        for (int k = 0; k < HDIM; ++k) {
            const float vk = Va_w[k], wk = wq[k];
            a0 = fmaf(vk, tanhf(wk + u0[k]), a0);
            a1 = fmaf(vk, tanhf(wk + u1[k]), a1);
        }
        sc0 = a0 + vab; sc1 = a1 + vab;
    }

    // softmax over 256 scores
    red[i] = fmaxf(sc0, sc1);
    __syncthreads();
    for (int off = 64; off > 0; off >>= 1) {
        if (i < off) red[i] = fmaxf(red[i], red[i + off]);
        __syncthreads();
    }
    const float mx = red[0];
    __syncthreads();
    const float e0 = __expf(sc0 - mx), e1 = __expf(sc1 - mx);
    wsm[i] = e0; wsm[i + 128] = e1;
    red[i] = e0 + e1;
    __syncthreads();
    for (int off = 64; off > 0; off >>= 1) {
        if (i < off) red[i] += red[i + off];
        __syncthreads();
    }
    const float inv = 1.0f / red[0];

    // context[i] = inv * sum_s wsm[s] * enc[b,s,i]
    float ctx = 0.0f;
    const float* eb = enc + (long long)b * SLEN * HDIM + i;
#pragma unroll 4
    for (int s = 0; s < SLEN; ++s) ctx = fmaf(wsm[s], eb[s * HDIM], ctx);
    ctx *= inv;

    // x = [emb[tok]; context]
    const int tk = tgt[b * TLEN + t];
    xs[i]       = emb[(long long)tk * HDIM + i];
    xs[128 + i] = ctx;
    __syncthreads();

    // GRU gates (each thread owns rows i, i+128, i+256)
    float gir = b_ih[i], giz = b_ih[128 + i], gin = b_ih[256 + i];
    {
        const float* r0 = W_ih + (long long)i * 256;
        const float* r1 = W_ih + (long long)(128 + i) * 256;
        const float* r2 = W_ih + (long long)(256 + i) * 256;
#pragma unroll 4
        for (int k = 0; k < 256; ++k) {
            const float xv = xs[k];
            gir = fmaf(r0[k], xv, gir);
            giz = fmaf(r1[k], xv, giz);
            gin = fmaf(r2[k], xv, gin);
        }
    }
    float ghr = b_hh[i], ghz = b_hh[128 + i], ghn = b_hh[256 + i];
    {
        const float* r0 = W_hh + (long long)i * HDIM;
        const float* r1 = W_hh + (long long)(128 + i) * HDIM;
        const float* r2 = W_hh + (long long)(256 + i) * HDIM;
#pragma unroll 4
        for (int k = 0; k < HDIM; ++k) {
            const float hv = q[k];
            ghr = fmaf(r0[k], hv, ghr);
            ghz = fmaf(r1[k], hv, ghz);
            ghn = fmaf(r2[k], hv, ghn);
        }
    }
    const float r = sigmoidf_(gir + ghr);
    const float z = sigmoidf_(giz + ghz);
    const float n = tanhf(gin + r * ghn);
    const float hn = (1.0f - z) * n + z * hi;

    h_next[b * HDIM + i] = hn;
    h_bf16[b * HDIM + i] = (__bf16)hn;
    if (hfin_out) hfin_out[b * HDIM + i] = hn;
    if (attn_out) {
        attn_out[b * SLEN + i]       = wsm[i] * inv;
        attn_out[b * SLEN + 128 + i] = wsm[i + 128] * inv;
    }
}

// ---------------------------------------------------------------------------
// Logits GEMM: out[m, n] = h_bf16[m,:128] . out_w_bf16[n,:128] + out_b[n]
// One N-tile per wave; each wave computes BOTH 16-row M-tiles (full M=32)
// against a single shared B fragment -> out_w (the dominant L2 traffic) is
// read exactly once per step. 8 x v_wmma_f32_16x16x32_bf16 per wave.
// Grid: 250 blocks x 256 threads = 2000 waves = 2000 N-tiles.
// ---------------------------------------------------------------------------
__global__ void k_logits(const __bf16* __restrict__ hbf, const __bf16* __restrict__ wbf,
                         const float* __restrict__ out_b, float* __restrict__ out) {
    const int tid   = threadIdx.x;
    const int lane  = tid & 31;
    const int wave  = tid >> 5;
    const int tileN = blockIdx.x * 8 + wave;   // 0..1999
    const int half  = lane >> 4;
    const int lm    = lane & 15;

    const int n = tileN * 16 + lm;

    union { bf16x16 v; uint4 q[2]; } A0, A1, Bm;
    union { floatx8 v; float f[8]; } C0, C1;

    const float bias = out_b[n];
#pragma unroll
    for (int r = 0; r < 8; ++r) { C0.f[r] = bias; C1.f[r] = bias; }

    const __bf16* a0row = hbf + lm * HDIM;                // M rows 0..15
    const __bf16* a1row = hbf + (16 + lm) * HDIM;         // M rows 16..31
    const __bf16* brow  = wbf + (long long)n * HDIM;      // B = out_w^T: column n == out_w row n

#pragma unroll
    for (int kb = 0; kb < 4; ++kb) {
        const int k0 = kb * 32;
        // A 16x32 bf16 layout: lanes<16 hold K {0..7,16..23}, lanes>=16 hold K {8..15,24..31}
        A0.q[0] = *(const uint4*)(a0row + k0 + half * 8);
        A0.q[1] = *(const uint4*)(a0row + k0 + 16 + half * 8);
        A1.q[0] = *(const uint4*)(a1row + k0 + half * 8);
        A1.q[1] = *(const uint4*)(a1row + k0 + 16 + half * 8);
        // B 32x16 bf16 layout: lanes<16 hold K {0..7,16..23}, lanes>=16 hold K {8..15,24..31} (same as A)
        Bm.q[0] = *(const uint4*)(brow + k0 + half * 8);
        Bm.q[1] = *(const uint4*)(brow + k0 + 16 + half * 8);
        C0.v = __builtin_amdgcn_wmma_f32_16x16x32_bf16(false, A0.v, false, Bm.v,
                                                       (short)0, C0.v, false, false);
        C1.v = __builtin_amdgcn_wmma_f32_16x16x32_bf16(false, A1.v, false, Bm.v,
                                                       (short)0, C1.v, false, false);
    }

#pragma unroll
    for (int r = 0; r < 8; ++r) {
        const int mr = half * 8 + r;                      // C layout: VGPR r -> row r (+8 for upper lanes)
        out[(long long)mr * VOCAB + n]        = C0.f[r];
        out[(long long)(16 + mr) * VOCAB + n] = C1.f[r];
    }
}

// ---------------------------------------------------------------------------
extern "C" void kernel_launch(void* const* d_in, const int* in_sizes, int n_in,
                              void* d_out, int out_size, void* d_ws, size_t ws_size,
                              hipStream_t stream) {
    const float* enc  = (const float*)d_in[0];
    const float* h0   = (const float*)d_in[1];
    const int*   tgt  = (const int*)  d_in[2];
    const float* emb  = (const float*)d_in[3];
    const float* Wa_w = (const float*)d_in[4];
    const float* Wa_b = (const float*)d_in[5];
    const float* Ua_w = (const float*)d_in[6];
    const float* Ua_b = (const float*)d_in[7];
    const float* Va_w = (const float*)d_in[8];
    const float* Va_b = (const float*)d_in[9];
    const float* W_ih = (const float*)d_in[10];
    const float* b_ih = (const float*)d_in[11];
    const float* W_hh = (const float*)d_in[12];
    const float* b_hh = (const float*)d_in[13];
    const float* outw = (const float*)d_in[14];
    const float* outb = (const float*)d_in[15];
    float* out = (float*)d_out;

    // Output layout: logits (T,B,V) | h_final (1,B,H) | attn (B,1,S)
    const long long OFF_HFIN = (long long)TLEN * BATCH * VOCAB;     // 131,072,000
    const long long OFF_ATTN = OFF_HFIN + BATCH * HDIM;             // +4096

    // Workspace carve-up
    char* ws = (char*)d_ws;
    float*  uk  = (float*)ws;                                       ws += (size_t)BATCH * SLEN * HDIM * 4;
    __bf16* wbf = (__bf16*)ws;                                      ws += (size_t)VOCAB * HDIM * 2;
    float*  hA  = (float*)ws;                                       ws += (size_t)BATCH * HDIM * 4;
    float*  hB  = (float*)ws;                                       ws += (size_t)BATCH * HDIM * 4;
    __bf16* hbf = (__bf16*)ws;

    // One-time prep
    {
        const int n = VOCAB * HDIM;
        k_f32_to_bf16<<<(n + 255) / 256, 256, 0, stream>>>(outw, wbf, n);
    }
    {
        const int n = BATCH * VOCAB;   // decoder_outputs[0] = zeros
        k_zero_f32<<<(n + 255) / 256, 256, 0, stream>>>(out, n);
    }
    {
        const int n = BATCH * HDIM;
        k_copy_f32<<<(n + 255) / 256, 256, 0, stream>>>(h0, hA, n);
    }
    k_uk<<<BATCH * SLEN, HDIM, 0, stream>>>(enc, Ua_w, Ua_b, uk);

    // Sequential decode: 127 steps, 2 kernels each
    for (int t = 0; t < TLEN - 1; ++t) {
        const float* hp = (t & 1) ? hB : hA;
        float*       hn = (t & 1) ? hA : hB;
        const bool last = (t == TLEN - 2);
        k_step<<<BATCH, HDIM, 0, stream>>>(enc, uk, tgt, emb, Wa_w, Wa_b, Va_w, Va_b,
                                           W_ih, b_ih, W_hh, b_hh, hp, hn, hbf,
                                           last ? (out + OFF_ATTN) : nullptr,
                                           last ? (out + OFF_HFIN) : nullptr, t);
        k_logits<<<250, 256, 0, stream>>>(hbf, wbf, outb,
                                          out + (long long)(t + 1) * BATCH * VOCAB);
    }
}
